// SelfAttention_76725295776159
// MI455X (gfx1250) — compile-verified, matrix operand order checked by reading, not run on hardware
//
#include <hip/hip_runtime.h>
#include <hip/hip_bf16.h>

// Problem constants (match reference)
#define B_    4
#define S_    2048
#define IN_   256
#define EMB_  512
#define H_    8
#define D_    64
#define FF_   1024
#define MROWS (B_*S_)   // 8192

typedef _Float16 h8   __attribute__((ext_vector_type(8)));
typedef _Float16 v16h __attribute__((ext_vector_type(16)));
typedef float    v8f  __attribute__((ext_vector_type(8)));

union HF { v16h v; h8 p[2]; };

// ---------------------------------------------------------------------------
// Conversions / transposes
// ---------------------------------------------------------------------------
__global__ __launch_bounds__(256) void k_f32_to_f16(const float* __restrict__ in,
                                                    _Float16* __restrict__ out, int n) {
  int i = blockIdx.x * 256 + threadIdx.x;
  if (i < n) out[i] = (_Float16)in[i];
}

// W is KxN row-major (fp32); Wt becomes NxK row-major (f16) so WMMA B-operand
// loads are contiguous per lane.
__global__ __launch_bounds__(256) void k_transpose_w(const float* __restrict__ W,
                                                     _Float16* __restrict__ Wt,
                                                     int K, int N) {
  int t = blockIdx.x * 256 + threadIdx.x;
  if (t >= K * N) return;
  int k = t % K, n = t / K;
  Wt[t] = (_Float16)W[(size_t)k * N + n];
}

// Vh [b*S+s][EMB] (f16) -> Vt [((b*H+h)*D+d)*S + s] (f16)
__global__ __launch_bounds__(256) void k_vtrans(const _Float16* __restrict__ Vh,
                                                _Float16* __restrict__ Vt) {
  int t = blockIdx.x * 256 + threadIdx.x;
  if (t >= B_ * H_ * D_ * S_) return;
  int s = t & (S_ - 1);
  int d = (t >> 11) & (D_ - 1);
  int h = (t >> 17) & (H_ - 1);
  int b = t >> 20;
  Vt[t] = Vh[((size_t)b * S_ + s) * EMB_ + h * D_ + d];
}

// ---------------------------------------------------------------------------
// Wave-level WMMA GEMM: C(MxN) = A(MxK,f16 rowmajor) * Bt(NxK,f16 rowmajor)^T
// + bias, optional ReLU. Each wave owns a 32x64 C macro-tile:
// 2 M-subtiles x 4 N-subtiles (8 f32 accumulators), sharing 2 A-fragments and
// 4 B-fragments per K-chunk -> 1.5 b128 loads per v_wmma instead of 4.
// ---------------------------------------------------------------------------
__global__ __launch_bounds__(128) void k_gemm16(const _Float16* __restrict__ A,
                                                const _Float16* __restrict__ Bt,
                                                const float* __restrict__ bias,
                                                float* __restrict__ Cf,
                                                _Float16* __restrict__ Ch,
                                                int M, int N, int K, int relu) {
  int tile = blockIdx.x * 4 + threadIdx.y;
  int tilesN = N >> 6;                 // 64-wide macro-tiles
  int total = (M >> 5) * tilesN;       // 32-tall macro-tiles
  if (tile >= total) return;
  int tm = tile / tilesN, tn = tile % tilesN;
  int lane = threadIdx.x;
  int r = lane & 15, hs = lane >> 4;

  const _Float16* arow0 = A + ((size_t)(tm * 32 + r)) * K;
  const _Float16* arow1 = arow0 + (size_t)16 * K;
  const _Float16* brow[4];
#pragma unroll
  for (int j = 0; j < 4; ++j)
    brow[j] = Bt + ((size_t)(tn * 64 + j * 16 + r)) * K;

  v8f acc[2][4];
#pragma unroll
  for (int mi = 0; mi < 2; ++mi)
#pragma unroll
    for (int j = 0; j < 4; ++j) acc[mi][j] = (v8f){};

  for (int k0 = 0; k0 < K; k0 += 32) {
    HF fa0, fa1, fb[4];
    fa0.p[0] = *(const h8*)(arow0 + k0 + hs * 8);
    fa0.p[1] = *(const h8*)(arow0 + k0 + 16 + hs * 8);
    fa1.p[0] = *(const h8*)(arow1 + k0 + hs * 8);
    fa1.p[1] = *(const h8*)(arow1 + k0 + 16 + hs * 8);
#pragma unroll
    for (int j = 0; j < 4; ++j) {
      fb[j].p[0] = *(const h8*)(brow[j] + k0 + hs * 8);
      fb[j].p[1] = *(const h8*)(brow[j] + k0 + 16 + hs * 8);
    }
    if (k0 + 32 < K) {                 // WGP-scope prefetch (all cache levels)
      __builtin_prefetch(arow0 + k0 + 32, 0, 3);
      __builtin_prefetch(arow1 + k0 + 32, 0, 3);
      __builtin_prefetch(brow[0] + k0 + 32, 0, 3);
      __builtin_prefetch(brow[2] + k0 + 32, 0, 3);
    }
#pragma unroll
    for (int j = 0; j < 4; ++j) {
      acc[0][j] = __builtin_amdgcn_wmma_f32_16x16x32_f16(false, fa0.v, false, fb[j].v,
                                                         (short)0, acc[0][j], false, false);
      acc[1][j] = __builtin_amdgcn_wmma_f32_16x16x32_f16(false, fa1.v, false, fb[j].v,
                                                         (short)0, acc[1][j], false, false);
    }
  }

#pragma unroll
  for (int j = 0; j < 4; ++j) {
    int n = tn * 64 + j * 16 + r;
    float bn = bias ? bias[n] : 0.0f;
#pragma unroll
    for (int mi = 0; mi < 2; ++mi) {
#pragma unroll
      for (int i = 0; i < 8; ++i) {
        int m = tm * 32 + mi * 16 + i + 8 * hs;  // C layout: row=i+8*(lane/16), col=lane&15
        float v = acc[mi][j][i] + bn;
        if (relu) v = fmaxf(v, 0.0f);
        size_t idx = (size_t)m * N + n;
        if (Cf) Cf[idx] = v;
        if (Ch) Ch[idx] = (_Float16)v;
      }
    }
  }
}

// ---------------------------------------------------------------------------
// Attention: one wave per (b, h, 16-query-row tile).
// Pass 1: row max of scaled QK^T. Pass 2: recompute, exp, row-sum, P@V.
// P relayout C->A goes through a 1KB LDS tile.
// ---------------------------------------------------------------------------
__global__ __launch_bounds__(32) void k_attn(const _Float16* __restrict__ Qh,
                                             const _Float16* __restrict__ Kh,
                                             const _Float16* __restrict__ Vt,
                                             float* __restrict__ ctx, float scale) {
  __shared__ _Float16 P[16][32];
  int lane = threadIdx.x;
  int r = lane & 15, hs = lane >> 4;
  int b = blockIdx.z, h = blockIdx.y, q0 = blockIdx.x * 16;

  // Q fragments (A operand, 16x64 split into two K=32 chunks)
  const _Float16* qbase = Qh + ((size_t)b * S_ + q0 + r) * EMB_ + h * D_;
  HF qa[2];
#pragma unroll
  for (int cc = 0; cc < 2; ++cc) {
    qa[cc].p[0] = *(const h8*)(qbase + cc * 32 + hs * 8);
    qa[cc].p[1] = *(const h8*)(qbase + cc * 32 + 16 + hs * 8);
  }

  // ---- pass 1: row max ----
  float mrow[8];
#pragma unroll
  for (int i = 0; i < 8; ++i) mrow[i] = -1e30f;
  for (int kt = 0; kt < S_ / 16; ++kt) {
    const _Float16* kbase = Kh + ((size_t)b * S_ + kt * 16 + r) * EMB_ + h * D_;
    HF kb0, kb1;
    kb0.p[0] = *(const h8*)(kbase + hs * 8);
    kb0.p[1] = *(const h8*)(kbase + 16 + hs * 8);
    kb1.p[0] = *(const h8*)(kbase + 32 + hs * 8);
    kb1.p[1] = *(const h8*)(kbase + 48 + hs * 8);
    v8f e = {};
    e = __builtin_amdgcn_wmma_f32_16x16x32_f16(false, qa[0].v, false, kb0.v, (short)0, e, false, false);
    e = __builtin_amdgcn_wmma_f32_16x16x32_f16(false, qa[1].v, false, kb1.v, (short)0, e, false, false);
#pragma unroll
    for (int i = 0; i < 8; ++i) mrow[i] = fmaxf(mrow[i], e[i]);
  }
#pragma unroll
  for (int i = 0; i < 8; ++i) {
    mrow[i] *= scale;  // scale > 0, commutes with max
#pragma unroll
    for (int m = 1; m < 16; m <<= 1)
      mrow[i] = fmaxf(mrow[i], __shfl_xor(mrow[i], m, 32));
  }

  // ---- pass 2: exp / sums / P@V ----
  float lrow[8];
  v8f o[4];
#pragma unroll
  for (int i = 0; i < 8; ++i) lrow[i] = 0.0f;
#pragma unroll
  for (int t = 0; t < 4; ++t) o[t] = (v8f){};

  for (int kk = 0; kk < S_ / 32; ++kk) {
    v8f e[2];
#pragma unroll
    for (int t = 0; t < 2; ++t) {
      const _Float16* kbase = Kh + ((size_t)b * S_ + kk * 32 + t * 16 + r) * EMB_ + h * D_;
      HF kb0, kb1;
      kb0.p[0] = *(const h8*)(kbase + hs * 8);
      kb0.p[1] = *(const h8*)(kbase + 16 + hs * 8);
      kb1.p[0] = *(const h8*)(kbase + 32 + hs * 8);
      kb1.p[1] = *(const h8*)(kbase + 48 + hs * 8);
      v8f et = {};
      et = __builtin_amdgcn_wmma_f32_16x16x32_f16(false, qa[0].v, false, kb0.v, (short)0, et, false, false);
      et = __builtin_amdgcn_wmma_f32_16x16x32_f16(false, qa[1].v, false, kb1.v, (short)0, et, false, false);
      e[t] = et;
    }
    __syncthreads();   // previous iteration's LDS reads complete
#pragma unroll
    for (int i = 0; i < 8; ++i) {
      float p0 = __expf(e[0][i] * scale - mrow[i]);
      float p1 = __expf(e[1][i] * scale - mrow[i]);
      lrow[i] += p0 + p1;
      P[i + 8 * hs][r]      = (_Float16)p0;   // C layout -> LDS
      P[i + 8 * hs][16 + r] = (_Float16)p1;
    }
    __syncthreads();
    HF pa;                                     // LDS -> A layout
    pa.p[0] = *(const h8*)&P[r][hs * 8];
    pa.p[1] = *(const h8*)&P[r][16 + hs * 8];
#pragma unroll
    for (int t = 0; t < 4; ++t) {
      const _Float16* vb = Vt + ((size_t)((b * H_ + h) * D_ + t * 16 + r)) * S_ + kk * 32 + hs * 8;
      HF vf;
      vf.p[0] = *(const h8*)vb;
      vf.p[1] = *(const h8*)(vb + 16);
      o[t] = __builtin_amdgcn_wmma_f32_16x16x32_f16(false, pa.v, false, vf.v, (short)0, o[t], false, false);
    }
  }

  float rinv[8];
#pragma unroll
  for (int i = 0; i < 8; ++i) {
#pragma unroll
    for (int m = 1; m < 16; m <<= 1)
      lrow[i] += __shfl_xor(lrow[i], m, 32);
    rinv[i] = 1.0f / lrow[i];
  }
#pragma unroll
  for (int t = 0; t < 4; ++t)
#pragma unroll
    for (int i = 0; i < 8; ++i) {
      int row = q0 + i + 8 * hs;
      ctx[((size_t)b * S_ + row) * EMB_ + h * D_ + t * 16 + r] = o[t][i] * rinv[i];
    }
}

// ---------------------------------------------------------------------------
// Fused residual + LayerNorm (one block per EMB=512 row; 256 threads x 2 elems)
// ---------------------------------------------------------------------------
__global__ __launch_bounds__(256) void k_ln(const float* __restrict__ xa,
                                            const float* __restrict__ xb,
                                            const float* __restrict__ g,
                                            const float* __restrict__ be,
                                            float* __restrict__ outf,
                                            _Float16* __restrict__ outh) {
  __shared__ float red[256];
  int row = blockIdx.x, t = threadIdx.x;
  const float* pa = xa + (size_t)row * EMB_;
  const float* pb = xb + (size_t)row * EMB_;
  float e0 = pa[t] + pb[t];
  float e1 = pa[t + 256] + pb[t + 256];
  red[t] = e0 + e1;
  __syncthreads();
  for (int m = 128; m > 0; m >>= 1) { if (t < m) red[t] += red[t + m]; __syncthreads(); }
  float mu = red[0] * (1.0f / EMB_);
  __syncthreads();
  float d0 = e0 - mu, d1 = e1 - mu;
  red[t] = d0 * d0 + d1 * d1;
  __syncthreads();
  for (int m = 128; m > 0; m >>= 1) { if (t < m) red[t] += red[t + m]; __syncthreads(); }
  float rs = rsqrtf(red[0] * (1.0f / EMB_) + 1e-5f);
  float y0 = d0 * rs * g[t] + be[t];
  float y1 = d1 * rs * g[t + 256] + be[t + 256];
  size_t base = (size_t)row * EMB_;
  outf[base + t] = y0;
  outf[base + t + 256] = y1;
  if (outh) { outh[base + t] = (_Float16)y0; outh[base + t + 256] = (_Float16)y1; }
}

// ---------------------------------------------------------------------------
// Host orchestration
// ---------------------------------------------------------------------------
extern "C" void kernel_launch(void* const* d_in, const int* in_sizes, int n_in,
                              void* d_out, int out_size, void* d_ws, size_t ws_size,
                              hipStream_t stream) {
  (void)in_sizes; (void)n_in; (void)out_size; (void)ws_size;
  const float* X    = (const float*)d_in[0];
  const float* qW1  = (const float*)d_in[1];  const float* qb1 = (const float*)d_in[2];
  const float* qW2  = (const float*)d_in[3];  const float* qb2 = (const float*)d_in[4];
  const float* kW1  = (const float*)d_in[5];  const float* kb1 = (const float*)d_in[6];
  const float* kW2  = (const float*)d_in[7];  const float* kb2 = (const float*)d_in[8];
  const float* vW1  = (const float*)d_in[9];  const float* vb1 = (const float*)d_in[10];
  const float* vW2  = (const float*)d_in[11]; const float* vb2 = (const float*)d_in[12];
  const float* fW1  = (const float*)d_in[13]; const float* fb1 = (const float*)d_in[14];
  const float* fW2  = (const float*)d_in[15]; const float* fb2 = (const float*)d_in[16];
  const float* ln1g = (const float*)d_in[17]; const float* ln1b = (const float*)d_in[18];
  const float* ln2g = (const float*)d_in[19]; const float* ln2b = (const float*)d_in[20];

  char* ws = (char*)d_ws;
  // ---- workspace plan (bytes, 256-aligned sizes) ----
  constexpr size_t SZ_XH   = (size_t)MROWS * IN_  * 2;   // 4 MB
  constexpr size_t SZ_W1T  = (size_t)EMB_  * IN_  * 2;   // 256 KB
  constexpr size_t SZ_W2T  = (size_t)EMB_  * EMB_ * 2;   // 512 KB
  constexpr size_t SZ_FW1T = (size_t)FF_   * EMB_ * 2;   // 1 MB
  constexpr size_t SZ_FW2T = (size_t)EMB_  * FF_  * 2;   // 1 MB
  constexpr size_t SZ_H16  = (size_t)MROWS * EMB_ * 2;   // 8 MB
  constexpr size_t SZ_F32  = (size_t)MROWS * EMB_ * 4;   // 16 MB

  size_t off = 0;
  auto take = [&](size_t sz) { size_t o = off; off += (sz + 255) & ~(size_t)255; return o; };
  const size_t O_XH   = take(SZ_XH);
  const size_t O_QW1T = take(SZ_W1T), O_KW1T = take(SZ_W1T), O_VW1T = take(SZ_W1T);
  const size_t O_QW2T = take(SZ_W2T), O_KW2T = take(SZ_W2T), O_VW2T = take(SZ_W2T);
  const size_t O_FW1T = take(SZ_FW1T), O_FW2T = take(SZ_FW2T);
  const size_t O_HH   = take(SZ_H16);       // q/k/v hidden (reused)
  const size_t O_QH   = take(SZ_H16);       // f16 Q  (O_HH..O_QH span reused as FF hidden)
  const size_t O_KH   = take(SZ_H16);
  const size_t O_VH   = take(SZ_H16);
  const size_t O_VT   = take(SZ_H16);
  const size_t O_X16  = take(SZ_H16);       // f16 LN1 output
  const size_t O_QF   = take(SZ_F32);       // f32 Q; later aliased by x (safe: per-thread RAW)
  const size_t O_CTX  = take(SZ_F32);       // ctx; later aliased by ff output
  const size_t O_FH   = O_HH;               // FF hidden (16 MB) reuses Hh+Qh region

  _Float16* Xh   = (_Float16*)(ws + O_XH);
  _Float16* Qh   = (_Float16*)(ws + O_QH);
  _Float16* Kh   = (_Float16*)(ws + O_KH);
  _Float16* Vh   = (_Float16*)(ws + O_VH);
  _Float16* Vt   = (_Float16*)(ws + O_VT);
  _Float16* Hh   = (_Float16*)(ws + O_HH);
  _Float16* Fh   = (_Float16*)(ws + O_FH);
  _Float16* x16  = (_Float16*)(ws + O_X16);
  float*    Qf   = (float*)(ws + O_QF);
  float*    ctx  = (float*)(ws + O_CTX);
  float*    xf   = Qf;     // alias
  float*    ff   = ctx;    // alias

  const float scale = 0.044194173824159216f;  // 1/sqrt(EMB)

  // conversions
  {
    int n = MROWS * IN_;
    k_f32_to_f16<<<(n + 255) / 256, 256, 0, stream>>>(X, Xh, n);
  }
  auto tw = [&](const float* W, size_t o, int K, int N) {
    int n = K * N;
    k_transpose_w<<<(n + 255) / 256, 256, 0, stream>>>(W, (_Float16*)(ws + o), K, N);
  };
  tw(qW1, O_QW1T, IN_, EMB_);  tw(kW1, O_KW1T, IN_, EMB_);  tw(vW1, O_VW1T, IN_, EMB_);
  tw(qW2, O_QW2T, EMB_, EMB_); tw(kW2, O_KW2T, EMB_, EMB_); tw(vW2, O_VW2T, EMB_, EMB_);
  tw(fW1, O_FW1T, EMB_, FF_);  tw(fW2, O_FW2T, FF_, EMB_);

  auto gemm = [&](const _Float16* A, size_t oBt, const float* bias, float* Cf,
                  _Float16* Ch, int M, int N, int K, int relu) {
    int tiles = (M >> 5) * (N >> 6);           // 32x64 macro-tiles
    dim3 blk(32, 4);
    k_gemm16<<<(tiles + 3) / 4, blk, 0, stream>>>(A, (const _Float16*)(ws + oBt),
                                                  bias, Cf, Ch, M, N, K, relu);
  };

  // Q/K/V projections (batch folded into M)
  gemm(Xh, O_QW1T, qb1, nullptr, Hh, MROWS, EMB_, IN_, 1);
  gemm(Hh, O_QW2T, qb2, Qf,      Qh, MROWS, EMB_, EMB_, 0);
  gemm(Xh, O_KW1T, kb1, nullptr, Hh, MROWS, EMB_, IN_, 1);
  gemm(Hh, O_KW2T, kb2, nullptr, Kh, MROWS, EMB_, EMB_, 0);
  gemm(Xh, O_VW1T, vb1, nullptr, Hh, MROWS, EMB_, IN_, 1);
  gemm(Hh, O_VW2T, vb2, nullptr, Vh, MROWS, EMB_, EMB_, 0);

  {
    int n = B_ * H_ * D_ * S_;
    k_vtrans<<<(n + 255) / 256, 256, 0, stream>>>(Vh, Vt);
  }

  k_attn<<<dim3(S_ / 16, H_, B_), 32, 0, stream>>>(Qh, Kh, Vt, ctx, scale);

  // x = LN(ctx + q)   (xf aliases Qf — per-thread read-before-write)
  k_ln<<<MROWS, 256, 0, stream>>>(ctx, Qf, ln1g, ln1b, xf, x16);

  // FFN
  gemm(x16, O_FW1T, fb1, nullptr, Fh, MROWS, FF_, EMB_, 1);
  gemm(Fh,  O_FW2T, fb2, ff, nullptr, MROWS, EMB_, FF_, 0);

  // out = LN(x + ff)
  k_ln<<<MROWS, 256, 0, stream>>>(xf, ff, ln2g, ln2b, (float*)d_out, nullptr);
}
